// Attention_4844723110629
// MI455X (gfx1250) — matrix operand order repaired and checked by reading, not compiled, not run on hardware
//
#include <hip/hip_runtime.h>
#include <math.h>

// Problem constants (match reference)
#define BB 16
#define TT 8192
#define DD 512
#define AA 128
#define ITER 4
#define ROWS_PER_WG (8 * 16 * ITER)   // 8 waves * 16 rows * ITER = 512
#define NTILES (TT / ROWS_PER_WG)     // 16
#define WFRAG_DWORDS (16 * 8 * 256)   // 16 kc * 8 nt * (32 lanes * 8 dwords) = 32768

typedef __attribute__((ext_vector_type(16))) __bf16 v16bf;
typedef __attribute__((ext_vector_type(8)))  float  v8f;
typedef __attribute__((ext_vector_type(4)))  int    v4i;

typedef __attribute__((address_space(1))) v4i* gptr_v4i;  // global (AS1)
typedef __attribute__((address_space(3))) v4i* sptr_v4i;  // LDS (AS3)

union Frag16 { unsigned u[8]; v16bf v; };

// Async global->LDS probe (gfx1250); fall back to plain copy if absent.
#if defined(__has_builtin)
#  if __has_builtin(__builtin_amdgcn_global_load_async_to_lds_b128)
#    define HAVE_ASYNC_LDS 1
#  endif
#endif
#ifndef HAVE_ASYNC_LDS
#  define HAVE_ASYNC_LDS 0
#endif

// Native V_TANH_F32 (gfx1250 trans op) if the builtin is exposed.
#if defined(__has_builtin)
#  if __has_builtin(__builtin_amdgcn_tanhf)
#    define fast_tanh(x) __builtin_amdgcn_tanhf(x)
#  endif
#endif
#ifndef fast_tanh
#  define fast_tanh(x) tanhf(x)
#endif

// fp32 pair -> packed bf16x2, round-to-nearest (ties up); 3 VALU ops total.
__device__ __forceinline__ unsigned pack_bf16(float a, float b) {
  unsigned ua = __float_as_uint(a) + 0x8000u;
  unsigned ub = __float_as_uint(b) + 0x8000u;
  // result = { ub[31:16], ua[31:16] }  (one v_perm_b32)
  return __builtin_amdgcn_perm(ub, ua, 0x07060302u);
}

// ---------------- kernel 0a: pq[b,a] = query[b,:] @ W_q[:,a] (tiny) --------
__global__ void attn_pq_kernel(const float* __restrict__ query,
                               const float* __restrict__ W_q,
                               float* __restrict__ pq) {
  const int b = blockIdx.x;
  const int a = threadIdx.x;
  float s = 0.f;
  for (int d = 0; d < DD; ++d) s += query[b * DD + d] * W_q[d * AA + a];
  pq[b * AA + a] = s;
}

// ------- kernel 0b: pack W_in into bf16 WMMA B-fragment layout (once) ------
// Layout: [kc 0..15][nt 0..7][lane 0..31][dword 0..7]
__global__ void attn_packW_kernel(const float* __restrict__ W_in,
                                  unsigned* __restrict__ wfrag) {
  const int idx  = blockIdx.x * 256 + threadIdx.x;  // 0..32767
  const int frag = idx >> 8;          // kc*8 + nt
  const int kc   = frag >> 3;
  const int nt   = frag & 7;
  const int r    = idx & 255;
  const int ls   = r >> 3;            // consuming lane 0..31
  const int i    = r & 7;             // dword within lane
  const int n    = nt * 16 + (ls & 15);
  const int kloc = (ls >> 4) << 4;    // lanes 16..31 hold K=16..31
  const int k    = kc * 32 + kloc + 2 * i;
  wfrag[idx] = pack_bf16(W_in[k * AA + n], W_in[(k + 1) * AA + n]);
}

// ---------------- kernel 1: fused scores + weighted accumulation ----------
__global__ __launch_bounds__(256)
void attn_fused_kernel(const float* __restrict__ values,
                       const unsigned* __restrict__ wfrag,
                       const float* __restrict__ v_w,
                       const float* __restrict__ v_b,
                       const float* __restrict__ pq,
                       float* __restrict__ part_ctx,
                       float* __restrict__ part_den) {
  // one K-quarter of B-fragments: 32 frags * 32 lanes * 8 dwords = 32 KB
  __shared__ __align__(16) unsigned ldsB[32 * 256];
  __shared__ float ldsCtx[DD];
  __shared__ float ldsDen;

  const int tid  = threadIdx.x;
  const int lane = tid & 31;
  const int wave = tid >> 5;
  const int tile = blockIdx.x;
  const int b    = blockIdx.y;
  const int t0   = tile * ROWS_PER_WG;

  ldsCtx[tid] = 0.f;
  ldsCtx[tid + 256] = 0.f;
  if (tid == 0) ldsDen = 0.f;

  const int ncol = lane & 15;          // N column / A-matrix row within tile
  const int klo  = (lane >> 4) * 16;   // A-fragment K sub-offset (0 or 16)

  float pqv[8], vw[8];
#pragma unroll
  for (int nt = 0; nt < 8; ++nt) {
    pqv[nt] = pq[b * AA + nt * 16 + ncol];
    vw[nt]  = v_w[nt * 16 + ncol];
  }
  const float vb = v_b[0];

  float ctx[16];
#pragma unroll
  for (int j = 0; j < 16; ++j) ctx[j] = 0.f;
  float wave_den = 0.f;

  for (int iter = 0; iter < ITER; ++iter) {
    const int m0 = t0 + (wave * ITER + iter) * 16;   // this wave's 16 rows

    v8f acc[8];
#pragma unroll
    for (int nt = 0; nt < 8; ++nt)
#pragma unroll
      for (int r = 0; r < 8; ++r) acc[nt][r] = 0.f;

    for (int q = 0; q < 4; ++q) {                    // K quarters of 128
      __syncthreads();   // previous quarter fully consumed
      // ---- stage quarter q: raw 32KB copy of pre-packed fragments ----
#if HAVE_ASYNC_LDS
      {
        const unsigned* gs = wfrag + q * (32 * 256);
        for (int idx = tid; idx < 2048; idx += 256) {
          gptr_v4i g = (gptr_v4i)(gs + idx * 4);
          sptr_v4i l = (sptr_v4i)(ldsB + idx * 4);
          __builtin_amdgcn_global_load_async_to_lds_b128(g, l, 0, 0);
        }
#  if __has_builtin(__builtin_amdgcn_s_wait_asynccnt)
        __builtin_amdgcn_s_wait_asynccnt(0);
#  else
        asm volatile("s_wait_asynccnt 0x0" ::: "memory");
#  endif
      }
#else
      {
        const v4i* gs = (const v4i*)(wfrag + q * (32 * 256));
        v4i*       ld = (v4i*)ldsB;
        for (int idx = tid; idx < 2048; idx += 256) ld[idx] = gs[idx];
      }
#endif
      __syncthreads();

#pragma unroll
      for (int kc_l = 0; kc_l < 4; ++kc_l) {
        const int kc = q * 4 + kc_l;
        // A fragment: 16 rows x 32 K of values, fp32 -> bf16 (per-lane)
        Frag16 af;
        const float* ap =
            values + ((long)b * TT + (m0 + ncol)) * (long)DD + kc * 32 + klo;
        const float4 f0 = *(const float4*)(ap);
        const float4 f1 = *(const float4*)(ap + 4);
        const float4 f2 = *(const float4*)(ap + 8);
        const float4 f3 = *(const float4*)(ap + 12);
        af.u[0] = pack_bf16(f0.x, f0.y); af.u[1] = pack_bf16(f0.z, f0.w);
        af.u[2] = pack_bf16(f1.x, f1.y); af.u[3] = pack_bf16(f1.z, f1.w);
        af.u[4] = pack_bf16(f2.x, f2.y); af.u[5] = pack_bf16(f2.z, f2.w);
        af.u[6] = pack_bf16(f3.x, f3.y); af.u[7] = pack_bf16(f3.z, f3.w);

#pragma unroll
        for (int nt = 0; nt < 8; ++nt) {
          Frag16 bf_;
          const v4i* bp = (const v4i*)&ldsB[(kc_l * 8 + nt) * 256 + lane * 8];
          const v4i b0 = bp[0], b1 = bp[1];
          bf_.u[0] = b0.x; bf_.u[1] = b0.y; bf_.u[2] = b0.z; bf_.u[3] = b0.w;
          bf_.u[4] = b1.x; bf_.u[5] = b1.y; bf_.u[6] = b1.z; bf_.u[7] = b1.w;
          acc[nt] = __builtin_amdgcn_wmma_f32_16x16x32_bf16(
              false, af.v, false, bf_.v, (short)0, acc[nt], false, false);
        }
      }
    }

    // ---- scores: lane L holds pv[row = r (+8 if L>=16)][N = nt*16+ncol] ----
    float sig[8];
#pragma unroll
    for (int r = 0; r < 8; ++r) {
      float s = 0.f;
#pragma unroll
      for (int nt = 0; nt < 8; ++nt)
        s += fast_tanh(acc[nt][r] + pqv[nt]) * vw[nt];
      // reduce over the 16 lanes of each half (N dimension)
      s += __shfl_xor(s, 1, 32);
      s += __shfl_xor(s, 2, 32);
      s += __shfl_xor(s, 4, 32);
      s += __shfl_xor(s, 8, 32);
      s += vb;
      sig[r] = 1.0f / (1.0f + __expf(-s));
    }
    float dsum = 0.f;
#pragma unroll
    for (int r = 0; r < 8; ++r) dsum += sig[r];
    dsum += __shfl_xor(dsum, 16, 32);   // combine rows 0-7 with rows 8-15
    wave_den += dsum;

    // ---- weighted accumulation of the same rows (cache-hot, float4) ----
    const float* vbase = values + ((long)b * TT + m0) * (long)DD;
#pragma unroll
    for (int m = 0; m < 16; ++m) {
      const float w = __shfl(sig[m & 7], (m < 8) ? 0 : 16, 32);
      const float4* vr4 = (const float4*)(vbase + (long)m * DD);
#pragma unroll
      for (int j = 0; j < 4; ++j) {
        const float4 v = vr4[lane + 32 * j];   // d = 4*lane + 128*j + c
        ctx[4 * j + 0] += w * v.x;
        ctx[4 * j + 1] += w * v.y;
        ctx[4 * j + 2] += w * v.z;
        ctx[4 * j + 3] += w * v.w;
      }
    }
  }

  // ---- combine 8 waves through LDS, emit per-tile partials ----
#pragma unroll
  for (int j = 0; j < 4; ++j)
#pragma unroll
    for (int c = 0; c < 4; ++c)
      atomicAdd(&ldsCtx[4 * lane + 128 * j + c], ctx[4 * j + c]);
  if (lane == 0) atomicAdd(&ldsDen, wave_den);
  __syncthreads();

  float* pc = part_ctx + ((long)b * NTILES + tile) * DD;
  pc[tid]       = ldsCtx[tid];
  pc[tid + 256] = ldsCtx[tid + 256];
  if (tid == 0) part_den[b * NTILES + tile] = ldsDen;
}

// ---------------- kernel 2: deterministic cross-tile reduction -------------
__global__ void attn_reduce_kernel(const float* __restrict__ part_ctx,
                                   const float* __restrict__ part_den,
                                   float* __restrict__ out) {
  const int b = blockIdx.x, tid = threadIdx.x;
  float den = 0.f;
  for (int i = 0; i < NTILES; ++i) den += part_den[b * NTILES + i];
  const float inv = 1.0f / den;
  for (int t = tid; t < DD; t += 256) {
    float s = 0.f;
    for (int i = 0; i < NTILES; ++i)
      s += part_ctx[((long)b * NTILES + i) * DD + t];
    out[b * DD + t] = s * inv;
  }
}

extern "C" void kernel_launch(void* const* d_in, const int* in_sizes, int n_in,
                              void* d_out, int out_size, void* d_ws, size_t ws_size,
                              hipStream_t stream) {
  const float* query  = (const float*)d_in[0];
  const float* values = (const float*)d_in[1];
  const float* W_in   = (const float*)d_in[2];
  const float* W_q    = (const float*)d_in[3];
  const float* v_w    = (const float*)d_in[4];
  const float* v_b    = (const float*)d_in[5];
  float* out = (float*)d_out;

  float* ws       = (float*)d_ws;
  float* pq       = ws;                        // BB*AA
  float* part_den = pq + BB * AA;              // BB*NTILES
  float* part_ctx = part_den + BB * NTILES;    // BB*NTILES*DD
  unsigned* wfrag = (unsigned*)(part_ctx + (long)BB * NTILES * DD); // 32768

  attn_pq_kernel<<<dim3(BB), dim3(AA), 0, stream>>>(query, W_q, pq);
  attn_packW_kernel<<<dim3(WFRAG_DWORDS / 256), dim3(256), 0, stream>>>(W_in, wfrag);
  attn_fused_kernel<<<dim3(NTILES, BB), dim3(256), 0, stream>>>(
      values, wfrag, v_w, v_b, pq, part_ctx, part_den);
  attn_reduce_kernel<<<dim3(BB), dim3(256), 0, stream>>>(part_ctx, part_den, out);
}
